// PyramidAttentionModel_80066780332686
// MI455X (gfx1250) — matrix operand order repaired and checked by reading, not compiled, not run on hardware
//
#include <hip/hip_runtime.h>

typedef __attribute__((ext_vector_type(16))) _Float16 v16h;
typedef __attribute__((ext_vector_type(8)))  _Float16 v8h;
typedef __attribute__((ext_vector_type(8)))  float    v8f;

#define NPIX 16384   // B*H*W = 16*32*32

// A-matrix (16x32 f16) per-lane K mapping (CDNA5 ISA 7.12.2):
// lanes 0-15: VGPR0..3 -> K=0..7,  VGPR4..7 -> K=16..23
// lanes16-31: VGPR0..3 -> K=8..15, VGPR4..7 -> K=24..31
// => fragment elems 0..7  = K = 8*half + (0..7)
//    fragment elems 8..15 = K = 16 + 8*half + (0..7)   (two contiguous 16B runs)
__device__ __forceinline__ int kmapA(int lane, int e) {
  int half = lane >> 4;
  int vg = e >> 1, pos = e & 1;
  int base = (vg < 4) ? (vg * 2) : (16 + (vg - 4) * 2);
  return base + pos + 8 * half;
}

// ---- pack x: x[16,16,32,32] f32 -> xg [g][c4][n] f16 (conv3 gather input),
//      and into padded channel-last rows of in5 (ch 64..67) / in7 (ch 128..131);
//      also zero the OOB zero-page.
__global__ __launch_bounds__(256)
void pack_x_kernel(const float* __restrict__ x, _Float16* __restrict__ xg,
                   _Float16* __restrict__ in5, _Float16* __restrict__ in7,
                   _Float16* __restrict__ zp) {
  int idx = blockIdx.x * 256 + threadIdx.x;        // 262144 = 16*16*1024
  if (idx < 512) zp[idx] = (_Float16)0.f;
  int hw = idx & 1023;
  int c  = (idx >> 10) & 15;
  int b  = idx >> 14;
  int g = c >> 2, c4 = c & 3;
  int n = b * 1024 + hw;
  _Float16 v = (_Float16)x[idx];
  xg[((size_t)(g * 4 + c4)) * NPIX + n] = v;
  int br0 = g * 2;
#pragma unroll
  for (int d = 0; d < 2; ++d) {
    in5[((size_t)(br0 + d) * NPIX + n) * 96  + 64  + c4] = v;
    in7[((size_t)(br0 + d) * NPIX + n) * 160 + 128 + c4] = v;
  }
}

// ---- conv3 weight swizzle (tiny; source K-order: cin*9 + widx) --------------
// dst: [br][mtile(4)][kit(2)][lane(32)][e(16)]
__global__ __launch_bounds__(256)
void pack_w3_kernel(const float* __restrict__ w, _Float16* __restrict__ dst) {
  int idx = blockIdx.x * 256 + threadIdx.x;        // 32768
  int e    = idx & 15;
  int lane = (idx >> 4) & 31;
  int kit  = (idx >> 9) & 1;
  int mtile = (idx >> 10) & 3;
  int br    = idx >> 12;
  int m = mtile * 16 + (lane & 15);
  int k = kit * 32 + kmapA(lane, e);
  _Float16 v = (_Float16)0.f;
  if (k < 36) v = (_Float16)w[((size_t)br * 64 + m) * 36 + k];
  dst[idx] = v;
}

// ---- conv5/7 weight repack to LINEAR shift-GEMM order -----------------------
// src row (contiguous): w[br][m][cin][widx]  (CIN_TOT*KS2 f32)
// dst row (contiguous): wlin[br*COUT+m][k],  k = widx*CINP + cin, f16, zero-pad
// One block per (br,m) row; source staged in LDS -> both sides coalesced.
__global__ __launch_bounds__(256)
void pack_w_lin(const float* __restrict__ w, _Float16* __restrict__ wlin,
                int CIN_TOT, int KS2, int CINP) {
  extern __shared__ float srow[];
  int row = blockIdx.x;                       // br*COUT + m
  const float* src = w + (size_t)row * CIN_TOT * KS2;
  int tot = CIN_TOT * KS2;
  for (int i = threadIdx.x; i < tot; i += 256) srow[i] = src[i];
  __syncthreads();
  int KTOT = KS2 * CINP;
  _Float16* drow = wlin + (size_t)row * KTOT;
  for (int k = threadIdx.x; k < KTOT; k += 256) {
    int widx = k / CINP;
    int cin  = k - widx * CINP;
    drow[k] = (cin < CIN_TOT) ? (_Float16)srow[cin * KS2 + widx] : (_Float16)0.f;
  }
}

// ---- conv3: 4ch -> 64, gather im2col (tiny layer), writes in5 rows ch 0..63 -
__global__ __launch_bounds__(256)
void conv3_wmma(const _Float16* __restrict__ xg,
                const _Float16* __restrict__ wSw,
                const float* __restrict__ bias,
                _Float16* __restrict__ out /* in5 */) {
  const int waveId = blockIdx.x * 8 + (threadIdx.x >> 5);  // 8192 waves
  const int lane = threadIdx.x & 31;
  const int ntile = waveId & 1023;
  const int br = waveId >> 10;
  const int grp = br >> 1;
  const int half = lane >> 4, col = lane & 15;
  const int n = ntile * 16 + col;
  const int b = n >> 10, h = (n >> 5) & 31, w = n & 31;
  v8f zero = {0.f,0.f,0.f,0.f,0.f,0.f,0.f,0.f};
  v8f acc[4];
#pragma unroll
  for (int i = 0; i < 4; ++i) acc[i] = zero;
  const _Float16* wb = wSw + (size_t)br * 8 * 512 + lane * 16;  // mtile stride 1024
  const _Float16* xgb = xg + (size_t)grp * 4 * NPIX;
#pragma unroll
  for (int kit = 0; kit < 2; ++kit) {
    v16h bfrag;
    const int kbase = kit * 32 + half * 8;
#pragma unroll
    for (int e = 0; e < 16; ++e) {
      int kk = kbase + e + ((e >= 8) ? 8 : 0);
      _Float16 v = (_Float16)0.f;
      if (kk < 36) {
        int cin = kk / 9;
        int r = kk - cin * 9;
        int hh = h + r / 3 - 1;
        int ww = w + r % 3 - 1;
        if (((unsigned)hh < 32u) && ((unsigned)ww < 32u))
          v = xgb[(size_t)cin * NPIX + (b << 10) + (hh << 5) + ww];
      }
      bfrag[e] = v;
    }
#pragma unroll
    for (int i = 0; i < 4; ++i) {
      v16h af = *(const v16h*)(wb + (size_t)(i * 2 + kit) * 512);
      acc[i] = __builtin_amdgcn_wmma_f32_16x16x32_f16(
          false, af, false, bfrag, (short)0, acc[i], false, false);
    }
  }
#pragma unroll
  for (int i = 0; i < 4; ++i) {
    int mbase = i * 16 + 8 * half;
    v8h pk;
#pragma unroll
    for (int j = 0; j < 8; ++j) {
      float val = acc[i][j] + bias[br * 64 + mbase + j];
      pk[j] = (_Float16)fmaxf(val, 0.f);
    }
    *(v8h*)(out + ((size_t)br * NPIX + n) * 96 + mbase) = pk;
  }
}

// ---- conv5/conv7: shift-GEMM WMMA, channel-last, contiguous loads -----------
// Block = 8 waves sharing (br, mg) -> A stream hits WGP$/L2.
// Each wave: MT=4 M-tiles x NT=2 N-tiles -> 8 WMMA per K-step.
// A fragments come straight from the LINEAR weight rows as two 16B loads.
template<int KS, int PAD, int CINP, int COUT, int MTILES, int OUTSTRIDE>
__global__ __launch_bounds__(256)
void conv_shift(const _Float16* __restrict__ in,    // [br][16384][CINP]
                const _Float16* __restrict__ zp,    // zero page
                const _Float16* __restrict__ wlin,  // [br*COUT+m][KTOT]
                const float* __restrict__ bias,
                _Float16* __restrict__ out) {       // [br][16384][OUTSTRIDE]
  constexpr int MG = MTILES / 4;
  constexpr int KS2 = KS * KS;
  constexpr int CSTEPS = CINP / 32;
  constexpr int KTOT = KS2 * CINP;
  const int wv = threadIdx.x >> 5;
  const int lane = threadIdx.x & 31;
  const int nb = blockIdx.x & 63;
  const int t = blockIdx.x >> 6;
  const int mg = t % MG;
  const int br = t / MG;
  const int half = lane >> 4, col = lane & 15;
  const int ntile0 = nb * 16 + wv * 2;
  const int n0 = ntile0 * 16 + col;

  v8f zero = {0.f,0.f,0.f,0.f,0.f,0.f,0.f,0.f};
  v8f acc[4][2];
#pragma unroll
  for (int i = 0; i < 4; ++i) { acc[i][0] = zero; acc[i][1] = zero; }

  // per-lane weight row pointers (m = col within each mtile), offset by half
  const _Float16* wrow[4];
#pragma unroll
  for (int i = 0; i < 4; ++i)
    wrow[i] = wlin + (size_t)(br * COUT + (mg * 4 + i) * 16 + col) * KTOT + 16 * half;

  const _Float16* inb = in + (size_t)br * NPIX * CINP;

#pragma unroll 1
  for (int widx = 0; widx < KS2; ++widx) {
    const int dy = widx / KS - PAD;
    const int dx = widx % KS - PAD;
    // per-pixel validity -> row base pointer (zero page when OOB)
    const _Float16* rp[2];
#pragma unroll
    for (int tt = 0; tt < 2; ++tt) {
      int nt = n0 + 16 * tt;
      int hh = ((nt >> 5) & 31) + dy;
      int ww = (nt & 31) + dx;
      bool valid = ((unsigned)hh < 32u) && ((unsigned)ww < 32u);
      rp[tt] = valid ? (inb + (size_t)(nt + dy * 32 + dx) * CINP) : zp;
    }
    const int ks0 = widx * CSTEPS;
    // prefetch next window's A chunk (gfx1250 global_prefetch)
    __builtin_prefetch(wrow[0] + (size_t)(ks0 + CSTEPS) * 32, 0, 1);
#pragma unroll
    for (int cs = 0; cs < CSTEPS; ++cs) {
      const int koff = (ks0 + cs) * 32;
      const int boff = cs * 32 + half * 16;
      v16h bf0 = *(const v16h*)(rp[0] + boff);
      v16h bf1 = *(const v16h*)(rp[1] + boff);
#pragma unroll
      for (int i = 0; i < 4; ++i) {
        v8h lo = *(const v8h*)(wrow[i] + koff);
        v8h hi = *(const v8h*)(wrow[i] + koff + 8);
        v16h af = __builtin_shufflevector(lo, hi, 0,1,2,3,4,5,6,7,8,9,10,11,12,13,14,15);
        acc[i][0] = __builtin_amdgcn_wmma_f32_16x16x32_f16(
            false, af, false, bf0, (short)0, acc[i][0], false, false);
        acc[i][1] = __builtin_amdgcn_wmma_f32_16x16x32_f16(
            false, af, false, bf1, (short)0, acc[i][1], false, false);
      }
    }
  }
  // epilogue: bias + ReLU, packed 16B channel-last stores
#pragma unroll
  for (int tt = 0; tt < 2; ++tt) {
    int nt = n0 + 16 * tt;
#pragma unroll
    for (int i = 0; i < 4; ++i) {
      int mbase = (mg * 4 + i) * 16 + 8 * half;
      v8h pk;
#pragma unroll
      for (int j = 0; j < 8; ++j) {
        float val = acc[i][tt][j] + bias[br * COUT + mbase + j];
        pk[j] = (_Float16)fmaxf(val, 0.f);
      }
      *(v8h*)(out + ((size_t)br * NPIX + nt) * OUTSTRIDE + mbase) = pk;
    }
  }
}

// ---- reductions: one channel per thread, fully coalesced --------------------
// S0 = sum_{h,w}, S1 = sum * cos(pi(w+.5)/32)  (K=2 DCT -> (0,0),(0,1))
__global__ void reduce_feat2(const _Float16* __restrict__ src, int rowStride,
                             int gcOff, float* __restrict__ gap_feat,
                             float* __restrict__ dct) {
  __shared__ float cvs[32];
  if (threadIdx.x < 32)
    cvs[threadIdx.x] = __cosf(3.14159265358979f * (threadIdx.x + 0.5f) / 32.0f);
  __syncthreads();
  int br = blockIdx.x >> 4;
  int b  = blockIdx.x & 15;
  const _Float16* p = src + ((size_t)br * NPIX + b * 1024) * rowStride + threadIdx.x;
  float s0 = 0.f, s1 = 0.f;
  for (int n = 0; n < 1024; ++n) {
    float v = (float)(*p);
    p += rowStride;
    s0 += v;
    s1 += v * cvs[n & 31];
  }
  int gc = br * 448 + gcOff + threadIdx.x;
  gap_feat[(size_t)b * 3584 + gc]    = s0 * (1.f / 1024.f);
  dct[(size_t)b * 7168 + 2 * gc]     = s0;
  dct[(size_t)b * 7168 + 2 * gc + 1] = s1;
}

// ---- LayerNorm statistics per sample ---------------------------------------
__global__ __launch_bounds__(256)
void ln_stats(const float* __restrict__ dct, float* __restrict__ stats) {
  __shared__ float ss[256], sq[256];
  int b = blockIdx.x;
  float s = 0.f, q = 0.f;
  for (int i = threadIdx.x; i < 7168; i += 256) {
    float v = dct[(size_t)b * 7168 + i];
    s += v; q += v * v;
  }
  ss[threadIdx.x] = s; sq[threadIdx.x] = q;
  __syncthreads();
  for (int off = 128; off > 0; off >>= 1) {
    if ((int)threadIdx.x < off) {
      ss[threadIdx.x] += ss[threadIdx.x + off];
      sq[threadIdx.x] += sq[threadIdx.x + off];
    }
    __syncthreads();
  }
  if (threadIdx.x == 0) {
    float mu = ss[0] / 7168.f;
    float var = sq[0] / 7168.f - mu * mu;
    stats[b * 2]     = mu;
    stats[b * 2 + 1] = rsqrtf(var + 1e-5f);
  }
}

// ---- both linear heads ------------------------------------------------------
__global__ __launch_bounds__(256)
void heads(const float* __restrict__ gap_feat, const float* __restrict__ dct,
           const float* __restrict__ stats,
           const float* __restrict__ Wg, const float* __restrict__ bg,
           const float* __restrict__ gamma, const float* __restrict__ beta,
           const float* __restrict__ Wd, const float* __restrict__ bd,
           float* __restrict__ outp) {
  int id = blockIdx.x * 8 + (threadIdx.x >> 5);   // 1920 waves
  int lane = threadIdx.x & 31;
  float s = 0.f;
  if (id < 960) {
    int b = id / 60, cls = id % 60;
    const float* f = gap_feat + (size_t)b * 3584;
    const float* wr = Wg + (size_t)cls * 3584;
    for (int i = lane; i < 3584; i += 32) s += f[i] * wr[i];
    for (int off = 16; off > 0; off >>= 1) s += __shfl_down(s, off, 32);
    if (lane == 0) outp[id] = s + bg[cls];
  } else {
    int t = id - 960;
    int b = t / 60, cls = t % 60;
    float mu = stats[b * 2], rstd = stats[b * 2 + 1];
    const float* f = dct + (size_t)b * 7168;
    const float* wr = Wd + (size_t)cls * 7168;
    for (int i = lane; i < 7168; i += 32) {
      float xn = (f[i] - mu) * rstd * gamma[i] + beta[i];
      s += xn * wr[i];
    }
    for (int off = 16; off > 0; off >>= 1) s += __shfl_down(s, off, 32);
    if (lane == 0) outp[960 + t] = s + bd[cls];
  }
}

extern "C" void kernel_launch(void* const* d_in, const int* in_sizes, int n_in,
                              void* d_out, int out_size, void* d_ws, size_t ws_size,
                              hipStream_t stream) {
  (void)in_sizes; (void)n_in; (void)out_size; (void)ws_size;
  const float* x  = (const float*)d_in[0];
  const float* w3 = (const float*)d_in[1];
  const float* b3 = (const float*)d_in[2];
  const float* w5 = (const float*)d_in[3];
  const float* b5 = (const float*)d_in[4];
  const float* w7 = (const float*)d_in[5];
  const float* b7 = (const float*)d_in[6];
  const float* Wg = (const float*)d_in[7];
  const float* bg = (const float*)d_in[8];
  const float* gm = (const float*)d_in[9];
  const float* bt = (const float*)d_in[10];
  const float* Wd = (const float*)d_in[11];
  const float* bd = (const float*)d_in[12];
  float* outp = (float*)d_out;

  char* ws = (char*)d_ws;
  // workspace layout (bytes), ~172.6 MB total, all 256-aligned
  _Float16* xg   = (_Float16*)(ws + 0);            //    524288
  _Float16* zp   = (_Float16*)(ws + 524288);       //      1024 (zero page)
  _Float16* w3s  = (_Float16*)(ws + 525312);       //     65536
  _Float16* w5l  = (_Float16*)(ws + 590848);       //   4915200  [1024][2400]
  _Float16* w7l  = (_Float16*)(ws + 5506048);      //  32112640  [2048][7840]
  _Float16* in5  = (_Float16*)(ws + 37618688);     //  25165824  [8][16384][96]
  _Float16* in7  = (_Float16*)(ws + 62784512);     //  41943040  [8][16384][160]
  _Float16* cAct = (_Float16*)(ws + 104727552);    //  67108864  [8][16384][256]
  float* gapF    = (float*)(ws + 171836416);       //    229376
  float* dct     = (float*)(ws + 172065792);       //    458752
  float* stats   = (float*)(ws + 172524544);       //       128

  // 1) pack input (xg + concat-x into in5/in7 rows) and repack weights
  pack_x_kernel<<<1024, 256, 0, stream>>>(x, xg, in5, in7, zp);
  pack_w3_kernel<<<128, 256, 0, stream>>>(w3, w3s);
  //          one block per (br,m) row      CIN_TOT KS2 CINP   (LDS = row of f32)
  pack_w_lin<<<1024, 256, 1700 * 4, stream>>>(w5, w5l, 68, 25,  96);
  pack_w_lin<<<2048, 256, 6468 * 4, stream>>>(w7, w7l, 132, 49, 160);

  // 2) convolutions via WMMA
  conv3_wmma<<<1024, 256, 0, stream>>>(xg, w3s, b3, in5);
  //          KS PAD CINP COUT MTILES OUTSTRIDE
  conv_shift<5, 2,  96, 128,  8, 160><<<1024, 256, 0, stream>>>(in5, zp, w5l, b5, in7);
  conv_shift<7, 3, 160, 256, 16, 256><<<2048, 256, 0, stream>>>(in7, zp, w7l, b7, cAct);

  // 3) fused GAP + DCT(K=2) reductions (one channel per thread, coalesced)
  reduce_feat2<<<128,  64, 0, stream>>>(in5,   96,   0, gapF, dct);
  reduce_feat2<<<128, 128, 0, stream>>>(in7,  160,  64, gapF, dct);
  reduce_feat2<<<128, 256, 0, stream>>>(cAct, 256, 192, gapF, dct);

  // 4) LayerNorm stats + both heads
  ln_stats<<<16, 256, 0, stream>>>(dct, stats);
  heads<<<240, 256, 0, stream>>>(gapF, dct, stats, Wg, bg, gm, bt, Wd, bd, outp);
}
